// NAryTreeLSTMCell_84189948936634
// MI455X (gfx1250) — compile-verified, matrix operand order checked by reading, not run on hardware
//
#include <hip/hip_runtime.h>
#include <hip/hip_bf16.h>
#include <math.h>

// Problem constants (from reference)
#define B_ROWS    16384
#define HD        512
#define LBL       128
#define ARG       1152          // 128 + 2*512
#define NG        5             // i, o, u, f0, f1

// GEMM tiling
#define BM        128
#define BN        32
#define BK        64            // two WMMA k-substeps per tile
#define KT64      (ARG / BK)    // 18
#define LDSP      72            // padded f16 row stride (144 B, conflict-free b128 reads)
#define XAREA     (BM * LDSP)          // 9216 f16
#define WAREA     (NG * BN * LDSP)     // 11520 f16
#define BUFSZ     (XAREA + WAREA)      // 20736 f16 per buffer
#define CHUNKS    9             // 16-byte copy chunks per thread per tile (2304/256)

typedef _Float16 v16h __attribute__((ext_vector_type(16)));
typedef _Float16 v8h  __attribute__((ext_vector_type(8)));
typedef _Float16 v4h  __attribute__((ext_vector_type(4)));
typedef float    v8f  __attribute__((ext_vector_type(8)));
typedef int      v4i_vs __attribute__((__vector_size__(16)));   // matches builtin param

union F16x16 { v16h v; v8h h[2]; };

// ---- CDNA5 async global->LDS path (guarded; falls back to direct copy) ----
#if __has_builtin(__builtin_amdgcn_global_load_async_to_lds_b128)
#define HAVE_ASYNC 1
#else
#define HAVE_ASYNC 0
#endif

#if HAVE_ASYNC
#if __has_builtin(__builtin_amdgcn_s_wait_asynccnt)
#define WAIT_ASYNC(n) __builtin_amdgcn_s_wait_asynccnt(n)
#else
#define WAIT_ASYNC(n) asm volatile("s_wait_asynccnt %0" :: "i"(n) : "memory")
#endif
#else
#define WAIT_ASYNC(n) ((void)0)
#endif

#if HAVE_ASYNC && __has_builtin(__builtin_amdgcn_s_barrier_signal) && \
    __has_builtin(__builtin_amdgcn_s_barrier_wait)
#define WG_BARRIER()                              \
    do {                                          \
        __builtin_amdgcn_s_barrier_signal(-1);    \
        __builtin_amdgcn_s_barrier_wait(-1);      \
    } while (0)
#else
#define WG_BARRIER() __syncthreads()
#endif

__device__ __forceinline__ void copy16_g2l(const _Float16* g, _Float16* l) {
#if HAVE_ASYNC
    __builtin_amdgcn_global_load_async_to_lds_b128(
        (__attribute__((address_space(1))) v4i_vs*)(_Float16*)g,
        (__attribute__((address_space(3))) v4i_vs*)l, 0, 0);
#else
    *(uint4*)l = *(const uint4*)g;
#endif
}

__device__ __forceinline__ float sigmoidf_fast(float x) {
    return 1.0f / (1.0f + __expf(-x));
}
__device__ __forceinline__ float tanhf_fast(float x) {
    // exact identity; fp error ~ulp(1), far below f16-GEMM noise
    return 1.0f - 2.0f / (__expf(2.0f * x) + 1.0f);
}

// ---------------------------------------------------------------------------
// Pack x = concat([label, h0, h1]) (B x 1152) into f16, vectorized by 4.
// ---------------------------------------------------------------------------
__global__ __launch_bounds__(256) void pack_x_kernel(
    const float* __restrict__ label,
    const float* __restrict__ chh,       // (2, B, 512)
    _Float16* __restrict__ X)
{
    int i   = blockIdx.x * 256 + threadIdx.x;        // over B * (ARG/4)
    int b   = i / (ARG / 4);
    int c4  = i - b * (ARG / 4);
    int col = c4 * 4;

    const float* src;
    if (col < LBL) {
        src = label + (size_t)b * LBL + col;
    } else {
        int r = col - LBL;
        int j = r >> 9;
        int c = r & 511;
        src = chh + ((size_t)j * B_ROWS + b) * HD + c;
    }
    float4 v = *(const float4*)src;
    v4h o = { (_Float16)v.x, (_Float16)v.y, (_Float16)v.z, (_Float16)v.w };
    *(v4h*)(X + (size_t)b * ARG + col) = o;
}

// ---------------------------------------------------------------------------
// Pack the 5 gate weight matrices into one (2560 x 1152) f16 matrix.
// ---------------------------------------------------------------------------
__global__ __launch_bounds__(256) void pack_w_kernel(
    const float* __restrict__ Wi, const float* __restrict__ Wo,
    const float* __restrict__ Wu, const float* __restrict__ Wfl,
    const float* __restrict__ Wfs,       // (2,2,512,512)
    _Float16* __restrict__ Wh)
{
    int i   = blockIdx.x * 256 + threadIdx.x;        // over (NG*HD) * (ARG/4)
    int r   = i / (ARG / 4);
    int c4  = i - r * (ARG / 4);
    int col = c4 * 4;
    int g   = r >> 9;
    int hh  = r & 511;

    const float* src;
    if (g == 0)      src = Wi + (size_t)hh * ARG + col;
    else if (g == 1) src = Wo + (size_t)hh * ARG + col;
    else if (g == 2) src = Wu + (size_t)hh * ARG + col;
    else {
        int k = g - 3;
        if (col < LBL) {
            src = Wfl + (size_t)hh * LBL + col;
        } else {
            int j = (col - LBL) >> 9;
            int c = (col - LBL) & 511;
            src = Wfs + (((size_t)(k * 2 + j) * HD + hh) * HD + c);
        }
    }
    float4 v = *(const float4*)src;
    v4h o = { (_Float16)v.x, (_Float16)v.y, (_Float16)v.z, (_Float16)v.w };
    *(v4h*)(Wh + (size_t)r * ARG + col) = o;
}

// ---------------------------------------------------------------------------
// Fused 5-gate WMMA GEMM + TreeLSTM epilogue.
// 256 threads = 8 waves; tile = 128 (batch) x 32 (hidden) x 5 gates, BK=64.
// Double-buffered LDS fed by GLOBAL_LOAD_ASYNC_TO_LDS_B128 (ASYNCcnt);
// B fragments software-pipelined through a 2-deep register buffer.
// ---------------------------------------------------------------------------
__global__ __launch_bounds__(256) void treelstm_gemm_kernel(
    const _Float16* __restrict__ X,      // (B, 1152) f16
    const _Float16* __restrict__ W,      // (2560, 1152) f16
    const float* __restrict__ b_i, const float* __restrict__ b_o,
    const float* __restrict__ b_u, const float* __restrict__ fbias,
    const float* __restrict__ cc,        // children_c (2, B, 512) f32
    float* __restrict__ out)             // [next_cell (B,512) | out (B,512)]
{
    __shared__ __align__(16) _Float16 smem[2 * BUFSZ];

    const int t       = threadIdx.x;
    const int lane    = t & 31;
    const int wv      = t >> 5;
    const int l15     = lane & 15;
    const int rowBase = blockIdx.y * BM;          // batch tile base
    const int nBase   = blockIdx.x * BN;          // hidden tile base

    // ---- uniform chunk map: 2304 x 16B chunks per tile, 9 per thread ----
    unsigned goff[CHUNKS];   // f16-element offset from X / W base (at k=0)
    int      lofs[CHUNKS];   // f16-element offset inside one LDS buffer
#pragma unroll
    for (int s = 0; s < CHUNKS; ++s) {
        if (s < 4) {
            int c    = t + 256 * s;
            int row  = c >> 3;
            int colc = (c & 7) * 8;
            goff[s]  = (unsigned)(rowBase + row) * ARG + colc;
            lofs[s]  = row * LDSP + colc;
        } else {
            int d    = t + 256 * (s - 4);
            int row  = d >> 3;               // 0..159
            int colc = (d & 7) * 8;
            int g    = row >> 5;
            int n    = row & 31;
            goff[s]  = (unsigned)(g * HD + nBase + n) * ARG + colc;
            lofs[s]  = XAREA + row * LDSP + colc;
        }
    }

    auto issueTile = [&](int kt, int buf) {
        const int k0 = kt * BK;
        _Float16* sb = smem + buf * BUFSZ;
#pragma unroll
        for (int s = 0; s < CHUNKS; ++s) {
            const _Float16* g = (s < 4 ? X : W) + goff[s] + k0;
            copy16_g2l(g, sb + lofs[s]);
        }
    };

    // ---- accumulators: 5 gates x 2 n-subtiles, f32 16x16 each = 80 VGPRs ----
    v8f acc[NG][2];
#pragma unroll
    for (int g = 0; g < NG; ++g)
#pragma unroll
        for (int nt = 0; nt < 2; ++nt) acc[g][nt] = (v8f)0.0f;

    // Fragment addressing per ISA 16-bit layout:
    // lanes 0-15: K 0-7 / 16-23, lanes 16-31: K 8-15 / 24-31.
    const int mrow = (wv << 4) + l15;
    const int kh   = (lane >> 4) << 3;            // 0 or 8
    // B row for pipeline step i (g = i>>1, nt = i&1):
    const int bRow0 = XAREA + l15 * LDSP + kh;    // + (g*BN + nt*16)*LDSP

    issueTile(0, 0);
#pragma unroll 2
    for (int kt = 0; kt < KT64; ++kt) {
        const int cur = kt & 1;
        WG_BARRIER();                       // buffer !cur now safe to overwrite
        if (kt + 1 < KT64) {
            issueTile(kt + 1, cur ^ 1);     // overlap next tile with compute
            WAIT_ASYNC(CHUNKS);             // in-order => tile kt complete
        } else {
            WAIT_ASYNC(0);
        }
        WG_BARRIER();                       // everyone's tile kt is in LDS

        const _Float16* sb = smem + cur * BUFSZ;
#pragma unroll
        for (int sub = 0; sub < 2; ++sub) {
            const int kk = sub * 32;
            F16x16 a;
            const _Float16* ap = sb + mrow * LDSP + kk + kh;
            a.h[0] = *(const v8h*)ap;
            a.h[1] = *(const v8h*)(ap + 16);

            // software-pipelined B fragments: load i+1 before WMMA i
            F16x16 b[2];
            {
                const _Float16* bp = sb + bRow0 + kk;          // step 0
                b[0].h[0] = *(const v8h*)bp;
                b[0].h[1] = *(const v8h*)(bp + 16);
            }
#pragma unroll
            for (int i = 0; i < 2 * NG; ++i) {
                if (i + 1 < 2 * NG) {
                    const int gn = ((i + 1) >> 1) * BN + ((i + 1) & 1) * 16;
                    const _Float16* bp = sb + bRow0 + gn * LDSP + kk;
                    b[(i + 1) & 1].h[0] = *(const v8h*)bp;
                    b[(i + 1) & 1].h[1] = *(const v8h*)(bp + 16);
                }
                acc[i >> 1][i & 1] = __builtin_amdgcn_wmma_f32_16x16x32_f16(
                    false, a.v, false, b[i & 1].v, (short)0, acc[i >> 1][i & 1],
                    false, false);
            }
        }
    }

    // ---- fused TreeLSTM epilogue ----
    // C/D layout: lane l holds N = l%16 (+16*nt); VGPR v holds M = v + 8*(l>=16).
    const int mOff = (wv << 4) + ((lane >> 4) << 3);
#pragma unroll
    for (int nt = 0; nt < 2; ++nt) {
        const int   h   = nBase + nt * 16 + l15;
        const float biv = b_i[h];
        const float bov = b_o[h];
        const float buv = b_u[h];
        const float fbv = fbias[h];
#pragma unroll
        for (int v = 0; v < 8; ++v) {
            const int   bIdx  = rowBase + mOff + v;
            const float ig    = sigmoidf_fast(acc[0][nt][v] + biv);
            const float og    = sigmoidf_fast(acc[1][nt][v] + bov);
            const float ug    = tanhf_fast(acc[2][nt][v] + buv);
            const float f0    = sigmoidf_fast(acc[3][nt][v]);
            const float f1    = sigmoidf_fast(acc[4][nt][v]);
            const float c0    = cc[(size_t)bIdx * HD + h];
            const float c1    = cc[((size_t)B_ROWS + bIdx) * HD + h];
            const float ncell = ig * ug + f0 * c0 + f1 * c1 + fbv * (c0 + c1);
            out[(size_t)bIdx * HD + h]            = ncell;
            out[((size_t)B_ROWS + bIdx) * HD + h] = tanhf_fast(og * ncell);
        }
    }
}

// ---------------------------------------------------------------------------
extern "C" void kernel_launch(void* const* d_in, const int* in_sizes, int n_in,
                              void* d_out, int out_size, void* d_ws, size_t ws_size,
                              hipStream_t stream) {
    (void)in_sizes; (void)n_in; (void)out_size; (void)ws_size;

    const float* label = (const float*)d_in[0];
    const float* chh   = (const float*)d_in[1];
    const float* chc   = (const float*)d_in[2];
    const float* Wi    = (const float*)d_in[3];
    const float* bi    = (const float*)d_in[4];
    const float* Wo    = (const float*)d_in[5];
    const float* bo    = (const float*)d_in[6];
    const float* Wu    = (const float*)d_in[7];
    const float* bu    = (const float*)d_in[8];
    const float* Wfl   = (const float*)d_in[9];
    const float* Wfs   = (const float*)d_in[10];
    const float* fb    = (const float*)d_in[11];

    _Float16* Xh = (_Float16*)d_ws;                       // 16384*1152 f16
    _Float16* Wh = Xh + (size_t)B_ROWS * ARG;             // 2560*1152 f16

    pack_x_kernel<<<(B_ROWS * (ARG / 4)) / 256, 256, 0, stream>>>(label, chh, Xh);
    pack_w_kernel<<<(NG * HD * (ARG / 4)) / 256, 256, 0, stream>>>(Wi, Wo, Wu,
                                                                   Wfl, Wfs, Wh);

    dim3 grid(HD / BN, B_ROWS / BM);                      // 16 x 128 = 2048 WGs
    treelstm_gemm_kernel<<<grid, 256, 0, stream>>>(Xh, Wh, bi, bo, bu, fb, chc,
                                                   (float*)d_out);
}
